// FCOS_26989574488510
// MI455X (gfx1250) — matrix operand order repaired and checked by reading, not compiled, not run on hardware
//
#include <hip/hip_runtime.h>
#include <hip/hip_bf16.h>

// ---------------- types ----------------
typedef __attribute__((ext_vector_type(16))) __bf16 v16bf;
typedef __attribute__((ext_vector_type(8)))  __bf16 v8bf;
typedef __attribute__((ext_vector_type(8)))  float  v8f;
typedef __attribute__((ext_vector_type(8)))  unsigned short u16x8;

__device__ __forceinline__ unsigned short f2bf(float f) {
    union { float f; unsigned u; } v; v.f = f;
    unsigned r = v.u + 0x7FFFu + ((v.u >> 16) & 1u);   // round-to-nearest-even
    return (unsigned short)(r >> 16);
}

// activations: bf16 NHWC [N*HW][256] ; packed weights: bf16 [9][OCpad][256]

// ---------------- NCHW f32 -> NHWC bf16 ----------------
__global__ __launch_bounds__(256)
void nchw_to_nhwc_bf16(const float* __restrict__ in, int HW, int P,
                       unsigned short* __restrict__ out) {
    int idx = blockIdx.x * 256 + threadIdx.x;     // one vec4 of channels
    if (idx >= P * 64) return;
    int pixel = idx >> 6;
    int c4    = (idx & 63) << 2;
    int n = pixel / HW;
    int p = pixel - n * HW;
    const float* src = in + ((size_t)n * 256 + c4) * HW + p;
    ushort4 o;
    o.x = f2bf(src[0]);
    o.y = f2bf(src[(size_t)HW]);
    o.z = f2bf(src[(size_t)2 * HW]);
    o.w = f2bf(src[(size_t)3 * HW]);
    *(ushort4*)&out[(size_t)pixel * 256 + c4] = o;
}

// ---------------- weight packing ----------------
__global__ __launch_bounds__(256)
void pack_tower_w(const float* __restrict__ w, unsigned short* __restrict__ wp, int total) {
    int idx = blockIdx.x * 256 + threadIdx.x;
    if (idx >= total) return;
    int ic  = idx & 255;
    int oc  = (idx >> 8) & 255;
    int tap = (idx >> 16) % 9;
    int l   = idx / (9 << 16);
    size_t src = (((size_t)l * 256 + oc) * 256 + ic) * 9 + tap;
    wp[idx] = f2bf(w[src]);
}

__global__ __launch_bounds__(256)
void pack_head_w(const float* __restrict__ cls, const float* __restrict__ bb,
                 const float* __restrict__ ctr, unsigned short* __restrict__ wp) {
    int idx = blockIdx.x * 256 + threadIdx.x;
    if (idx >= 9 * 128 * 256) return;
    int ic  = idx & 255;
    int oc  = (idx >> 8) & 127;
    int tap = idx >> 15;
    float v = 0.f;
    if (oc < 80)       v = cls[((size_t)oc * 256 + ic) * 9 + tap];
    else if (oc < 84)  v = bb[((size_t)(oc - 80) * 256 + ic) * 9 + tap];
    else if (oc == 84) v = ctr[(size_t)ic * 9 + tap];
    wp[idx] = f2bf(v);
}

__global__ void pack_head_bias(const float* __restrict__ clsb, const float* __restrict__ bbb,
                               const float* __restrict__ ctrb, float* __restrict__ hb) {
    int oc = threadIdx.x;
    float v = 0.f;
    if (oc < 80)       v = clsb[oc];
    else if (oc < 84)  v = bbb[oc - 80];
    else if (oc == 84) v = ctrb[0];
    hb[oc] = v;
}

// ---------------- implicit-GEMM 3x3 conv via bf16 WMMA ----------------
// Workgroup tile: 128 pixels x 128 oc; 8 waves = 4 px-tiles x 2 oc-tiles.
// Wave tile: 32 px x 64 oc -> 2 A-frags x 4 B-frags, 8 WMMAs / K-step.
// Double-buffered LDS, one barrier per K-step, global loads overlapped.
template <bool HEAD>
__global__ __launch_bounds__(256)
void conv3x3_wmma(const unsigned short* __restrict__ x,   // bf16 NHWC [P][256]
                  const unsigned short* __restrict__ wp,  // bf16 [9][OCpad][256]
                  const float* __restrict__ bias,         // [OCpad]
                  float* __restrict__ y,                  // tower: f32 NHWC [P][256]
                  float* __restrict__ out,                // head: [N][totalP][85]
                  int H, int W, int HW, int P, int OCpad,
                  int totalP, int levelOff, int ocLo, int ocHi) {
    const int tid  = threadIdx.x;
    const int wave = tid >> 5;
    const int lane = tid & 31;
    const int pxBlock = blockIdx.x * 128;
    const int ocBlock = blockIdx.y * 128;

    __shared__ unsigned short Alds[2][128 * 40];   // 128 px rows x (32 K + pad)
    __shared__ unsigned short Blds[2][128 * 40];   // 128 oc rows x (32 K + pad)

    // staging: 256 thr = 128 rows x 2 halves (16 bf16 = 32B each)
    const int srow  = tid >> 1;
    const int shalf = tid & 1;

    // A-side per-thread geometry (tap-independent precompute)
    const int  apx = pxBlock + srow;
    const bool apValid = apx < P;
    int an = 0, aoy = 0, aox = 0;
    if (apValid) { an = apx / HW; int rem = apx - an * HW; aoy = rem / W; aox = rem - aoy * W; }
    const long centerBase = apValid ? (long)((an * H + aoy) * W + aox) * 256 : 0;
    const long rowStride  = (long)W * 256;
    unsigned vmask = 0;
    if (apValid) {
#pragma unroll
        for (int t = 0; t < 9; ++t) {
            int iy = aoy + t / 3 - 1, ix = aox + t % 3 - 1;
            if ((unsigned)iy < (unsigned)H && (unsigned)ix < (unsigned)W) vmask |= 1u << t;
        }
    }
    const int aStageOff = srow * 40 + shalf * 16;            // LDS elements
    const long bRowOff  = (long)(ocBlock + srow) * 256 + shalf * 16;

    // fragment addressing (ISA 16-bit A 16x32 / B 32x16 layouts)
    const int lhalf  = lane >> 4;
    const int lmod   = lane & 15;
    const int pxTile = wave & 3;     // 32 px each
    const int ocTile = wave >> 2;    // 64 oc each
    const int aFr0 = (pxTile * 32 + lmod) * 40 + lhalf * 8;  // K {0-7,16-23}/{8-15,24-31}
    const int aFr1 = aFr0 + 16 * 40;

    v8f acc[2][4];
#pragma unroll
    for (int m = 0; m < 2; ++m)
#pragma unroll
        for (int j = 0; j < 4; ++j)
#pragma unroll
            for (int r = 0; r < 8; ++r) acc[m][j][r] = 0.0f;

    // chunk index -> global addresses (chunk kk: tap = kk/8, kc = kk%8)
    auto chunkAddr = [&](int kk, long& aAddr, bool& av, long& bAddr) {
        int tap = kk >> 3, kc = kk & 7;
        int dyi = tap / 3, dxi = tap - dyi * 3;              // 0..2
        av = (vmask >> tap) & 1u;
        aAddr = centerBase + (long)(dyi - 1) * rowStride + (long)(dxi - 1) * 256
              + kc * 32 + shalf * 16;
        bAddr = (long)tap * OCpad * 256 + bRowOff + kc * 32;
    };
    auto loadChunk = [&](long aAddr, bool av, long bAddr,
                         u16x8& a0, u16x8& a1, u16x8& b0, u16x8& b1) {
        if (av) {
            a0 = *(const u16x8*)&x[aAddr];
            a1 = *(const u16x8*)&x[aAddr + 8];
        } else {
#pragma unroll
            for (int i = 0; i < 8; ++i) { a0[i] = 0; a1[i] = 0; }
        }
        b0 = *(const u16x8*)&wp[bAddr];
        b1 = *(const u16x8*)&wp[bAddr + 8];
    };

    // prologue: stage chunk 0 into buffer 0
    {
        long aA, bA; bool av;
        chunkAddr(0, aA, av, bA);
        u16x8 a0, a1, b0, b1;
        loadChunk(aA, av, bA, a0, a1, b0, b1);
        *(u16x8*)&Alds[0][aStageOff]     = a0;
        *(u16x8*)&Alds[0][aStageOff + 8] = a1;
        *(u16x8*)&Blds[0][aStageOff]     = b0;
        *(u16x8*)&Blds[0][aStageOff + 8] = b1;
    }
    __syncthreads();

    for (int kk = 0; kk < 72; ++kk) {
        const int cur = kk & 1;
        u16x8 a0, a1, b0, b1;
        if (kk < 71) {                       // overlap next chunk's global loads
            long aA, bA; bool av;
            chunkAddr(kk + 1, aA, av, bA);
            loadChunk(aA, av, bA, a0, a1, b0, b1);
        }

        // compute on current buffer
        v8bf al0 = *(const v8bf*)&Alds[cur][aFr0];
        v8bf ah0 = *(const v8bf*)&Alds[cur][aFr0 + 16];
        v8bf al1 = *(const v8bf*)&Alds[cur][aFr1];
        v8bf ah1 = *(const v8bf*)&Alds[cur][aFr1 + 16];
        v16bf af0 = __builtin_shufflevector(al0, ah0,
            0, 1, 2, 3, 4, 5, 6, 7, 8, 9, 10, 11, 12, 13, 14, 15);
        v16bf af1 = __builtin_shufflevector(al1, ah1,
            0, 1, 2, 3, 4, 5, 6, 7, 8, 9, 10, 11, 12, 13, 14, 15);
#pragma unroll
        for (int j = 0; j < 4; ++j) {
            const int col = ocTile * 64 + j * 16 + lmod;
            v8bf bl = *(const v8bf*)&Blds[cur][col * 40 + lhalf * 8];
            v8bf bh = *(const v8bf*)&Blds[cur][col * 40 + lhalf * 8 + 16];
            v16bf bf = __builtin_shufflevector(bl, bh,
                0, 1, 2, 3, 4, 5, 6, 7, 8, 9, 10, 11, 12, 13, 14, 15);
            acc[0][j] = __builtin_amdgcn_wmma_f32_16x16x32_bf16(
                false, af0, false, bf, (short)0, acc[0][j], false, false);
            acc[1][j] = __builtin_amdgcn_wmma_f32_16x16x32_bf16(
                false, af1, false, bf, (short)0, acc[1][j], false, false);
        }

        if (kk < 71) {                       // stage next chunk into other buffer
            const int nxt = cur ^ 1;
            *(u16x8*)&Alds[nxt][aStageOff]     = a0;
            *(u16x8*)&Alds[nxt][aStageOff + 8] = a1;
            *(u16x8*)&Blds[nxt][aStageOff]     = b0;
            *(u16x8*)&Blds[nxt][aStageOff + 8] = b1;
            __syncthreads();
        }
    }

    // epilogue: D layout -> M = r + 8*lhalf, N = lane&15
#pragma unroll
    for (int m = 0; m < 2; ++m) {
#pragma unroll
        for (int j = 0; j < 4; ++j) {
            const int oc = ocBlock + ocTile * 64 + j * 16 + lmod;
            const float b = bias[oc];
#pragma unroll
            for (int r = 0; r < 8; ++r) {
                const int px = pxBlock + pxTile * 32 + m * 16 + r + (lhalf << 3);
                if (px < P) {
                    float val = acc[m][j][r] + b;
                    if (!HEAD) {
                        y[(size_t)px * 256 + oc] = val;
                    } else {
                        if (oc >= ocLo && oc < ocHi) {
                            if (oc >= 80 && oc < 84) val = fmaxf(val, 0.f);  // bbox ReLU
                            int n = px / HW;
                            int p = px - n * HW;
                            out[((size_t)n * totalP + levelOff + p) * 85 + oc] = val;
                        }
                    }
                }
            }
        }
    }
}

// ---------------- GroupNorm ----------------
__global__ __launch_bounds__(256)
void gn_stats(const float* __restrict__ y, int HW, float* __restrict__ stats) {
    const int n = blockIdx.x >> 4;
    const int g = blockIdx.x & 15;
    const int tid = threadIdx.x;
    float s = 0.f, ss = 0.f;
    for (int p = tid; p < HW; p += 256) {
        const float* base = y + ((size_t)(n * HW + p)) * 256 + g * 16;
#pragma unroll
        for (int c = 0; c < 16; c += 4) {
            float4 v = *(const float4*)(base + c);
            s  += v.x + v.y + v.z + v.w;
            ss += v.x * v.x + v.y * v.y + v.z * v.z + v.w * v.w;
        }
    }
    __shared__ float sh[512];
    sh[tid] = s; sh[256 + tid] = ss;
    __syncthreads();
    for (int o = 128; o > 0; o >>= 1) {
        if (tid < o) { sh[tid] += sh[tid + o]; sh[256 + tid] += sh[256 + tid + o]; }
        __syncthreads();
    }
    if (tid == 0) {
        float cnt = (float)HW * 16.f;
        float m = sh[0] / cnt;
        float var = sh[256] / cnt - m * m;
        stats[blockIdx.x * 2]     = m;
        stats[blockIdx.x * 2 + 1] = rsqrtf(var + 1e-5f);
    }
}

__global__ __launch_bounds__(256)
void gn_apply_relu(const float* __restrict__ y, const float* __restrict__ stats,
                   const float* __restrict__ gamma, const float* __restrict__ beta,
                   int HW, int P, unsigned short* __restrict__ xout) {
    int idx = blockIdx.x * 256 + threadIdx.x;     // one vec4 of channels
    if (idx >= P * 64) return;
    int pixel = idx >> 6;
    int c4    = (idx & 63) << 2;
    int n = pixel / HW;
    int g = c4 >> 4;
    float m  = stats[(n * 16 + g) * 2];
    float rs = stats[(n * 16 + g) * 2 + 1];
    const float4 v = *(const float4*)&y[(size_t)pixel * 256 + c4];
    ushort4 o;
    o.x = f2bf(fmaxf((v.x - m) * rs * gamma[c4 + 0] + beta[c4 + 0], 0.f));
    o.y = f2bf(fmaxf((v.y - m) * rs * gamma[c4 + 1] + beta[c4 + 1], 0.f));
    o.z = f2bf(fmaxf((v.z - m) * rs * gamma[c4 + 2] + beta[c4 + 2], 0.f));
    o.w = f2bf(fmaxf((v.w - m) * rs * gamma[c4 + 3] + beta[c4 + 3], 0.f));
    *(ushort4*)&xout[(size_t)pixel * 256 + c4] = o;
}

// ---------------- host orchestration ----------------
extern "C" void kernel_launch(void* const* d_in, const int* in_sizes, int n_in,
                              void* d_out, int out_size, void* d_ws, size_t ws_size,
                              hipStream_t stream) {
    (void)in_sizes; (void)n_in; (void)out_size; (void)ws_size;
    static const int HS[5] = {100, 50, 25, 13, 7};
    static const int WS[5] = {128, 64, 32, 16, 8};
    const int NB = 2, C = 256, TOTALP = 17064;

    const float* feats[5];
    for (int i = 0; i < 5; ++i) feats[i] = (const float*)d_in[i];
    const float* cls_conv_w = (const float*)d_in[5];
    const float* cls_conv_b = (const float*)d_in[6];
    const float* cls_gn_g   = (const float*)d_in[7];
    const float* cls_gn_b   = (const float*)d_in[8];
    const float* cls_log_w  = (const float*)d_in[9];
    const float* cls_log_b  = (const float*)d_in[10];
    const float* reg_conv_w = (const float*)d_in[11];
    const float* reg_conv_b = (const float*)d_in[12];
    const float* reg_gn_g   = (const float*)d_in[13];
    const float* reg_gn_b   = (const float*)d_in[14];
    const float* bbox_w     = (const float*)d_in[15];
    const float* bbox_b     = (const float*)d_in[16];
    const float* ctr_w      = (const float*)d_in[17];
    const float* ctr_b      = (const float*)d_in[18];
    float* out = (float*)d_out;

    // workspace carve-out
    char* ws = (char*)d_ws;
    size_t off = 0;
    auto carve = [&](size_t bytes) { char* p = ws + off; off += (bytes + 255) & ~(size_t)255; return p; };
    const size_t maxP = (size_t)NB * HS[0] * WS[0];
    unsigned short* featbuf  = (unsigned short*)carve(maxP * C * 2);
    unsigned short* xb0      = (unsigned short*)carve(maxP * C * 2);
    unsigned short* xb1      = (unsigned short*)carve(maxP * C * 2);
    float*          convout  = (float*)carve(maxP * C * 4);
    unsigned short* packCls  = (unsigned short*)carve((size_t)3 * 9 * 256 * 256 * 2);
    unsigned short* packReg  = (unsigned short*)carve((size_t)3 * 9 * 256 * 256 * 2);
    unsigned short* packHead = (unsigned short*)carve((size_t)9 * 128 * 256 * 2);
    float*          headBias = (float*)carve(128 * 4);
    float*          stats    = (float*)carve(32 * 2 * 4);

    // pack weights (deterministic, every call)
    {
        int tot = 3 * 9 * 256 * 256;
        pack_tower_w<<<(tot + 255) / 256, 256, 0, stream>>>(cls_conv_w, packCls, tot);
        pack_tower_w<<<(tot + 255) / 256, 256, 0, stream>>>(reg_conv_w, packReg, tot);
        int htot = 9 * 128 * 256;
        pack_head_w<<<(htot + 255) / 256, 256, 0, stream>>>(cls_log_w, bbox_w, ctr_w, packHead);
        pack_head_bias<<<1, 128, 0, stream>>>(cls_log_b, bbox_b, ctr_b, headBias);
    }

    int levelOff = 0;
    for (int l = 0; l < 5; ++l) {
        const int H = HS[l], W = WS[l], HW = H * W, P = NB * HW;
        const int vecCnt = P * 64;
        nchw_to_nhwc_bf16<<<(vecCnt + 255) / 256, 256, 0, stream>>>(feats[l], HW, P, featbuf);

        for (int tower = 0; tower < 2; ++tower) {
            const unsigned short* pack = tower ? packReg : packCls;
            const float* cb = tower ? reg_conv_b : cls_conv_b;
            const float* gg = tower ? reg_gn_g : cls_gn_g;
            const float* gb = tower ? reg_gn_b : cls_gn_b;
            const unsigned short* cur = featbuf;
            for (int i = 0; i < 3; ++i) {
                dim3 grid((P + 127) / 128, 2);
                conv3x3_wmma<false><<<grid, 256, 0, stream>>>(
                    cur, pack + (size_t)i * 9 * 256 * 256, cb + i * 256,
                    convout, nullptr, H, W, HW, P, 256, 0, 0, 0, 0);
                gn_stats<<<32, 256, 0, stream>>>(convout, HW, stats);
                unsigned short* nxt = (i & 1) ? xb1 : xb0;
                gn_apply_relu<<<(vecCnt + 255) / 256, 256, 0, stream>>>(
                    convout, stats, gg + i * 256, gb + i * 256, HW, P, nxt);
                cur = nxt;
            }
            // heads: cls tower -> channels [0,80); reg tower -> [80,85)
            const int ocLo = tower ? 80 : 0;
            const int ocHi = tower ? 85 : 80;
            dim3 hgrid((P + 127) / 128, 1);
            conv3x3_wmma<true><<<hgrid, 256, 0, stream>>>(
                cur, packHead, headBias, nullptr, out,
                H, W, HW, P, 128, TOTALP, levelOff, ocLo, ocHi);
        }
        levelOff += HW;
    }
}
